// GPT5MoELayer_41824391528975
// MI455X (gfx1250) — matrix operand order repaired and checked by reading, not compiled, not run on hardware
//
#include <hip/hip_runtime.h>

#define HDIM 1024
#define FDIM 2048
#define NEXP 32
#define TTOK 4096
#define TKE  (TTOK * 2)
#define CAPE 512

typedef __attribute__((ext_vector_type(16))) __bf16 v16bf;
typedef __attribute__((ext_vector_type(8)))  float  v8f;

// f32 -> bf16 (round-to-nearest; differs from RNE only on exact ties)
__device__ __forceinline__ unsigned short f2bf(float f) {
    union { float f; unsigned int u; } c; c.f = f;
    return (unsigned short)((c.u + 0x8000u) >> 16);
}

// Pack two f32 -> bf16x2 in one dword: two adds + one v_perm_b32.
__device__ __forceinline__ unsigned int pkbf(float lo, float hi) {
    union { float f; unsigned int u; } a, b;
    a.f = lo; b.f = hi;
    unsigned int ul = a.u + 0x8000u;
    unsigned int uh = b.u + 0x8000u;
    // result bytes: [1:0] = ul[3:2], [3:2] = uh[3:2]
    return __builtin_amdgcn_perm(uh, ul, 0x07060302u);
}

__device__ __forceinline__ void wait_async0() {
#if __has_builtin(__builtin_amdgcn_s_wait_asynccnt)
    __builtin_amdgcn_s_wait_asynccnt(0);
#else
    asm volatile("s_wait_asynccnt 0" ::: "memory");
#endif
}

__device__ __forceinline__ v8f wmma_bf16(v16bf a, v16bf b, v8f c) {
    return __builtin_amdgcn_wmma_f32_16x16x32_bf16(false, a, false, b, (short)0, c, false, false);
}

// ---------------------------------------------------------------- zero
__global__ void moe_zero(float* __restrict__ p) {
    p[threadIdx.x] = 0.0f;   // probs_sum[32] + top1_cnt[32], contiguous
}

// ---------------------------------------------------------------- router
// 1 wave32 per token, lane == expert index.
__global__ __launch_bounds__(128) void moe_router(
    const float* __restrict__ x, const float* __restrict__ Wr, const float* __restrict__ br,
    int* __restrict__ idx_flat, float* __restrict__ w_flat,
    float* __restrict__ probs_sum, float* __restrict__ top1_cnt)
{
    int wid  = threadIdx.x >> 5;
    int lane = threadIdx.x & 31;
    int t = blockIdx.x * 4 + wid;

    const float4* xr = (const float4*)(x + (size_t)t * HDIM);
    const float4* wr = (const float4*)(Wr + (size_t)lane * HDIM);
    float acc = 0.0f;
    for (int i = 0; i < HDIM / 4; ++i) {
        float4 a = xr[i], b = wr[i];
        acc += a.x * b.x + a.y * b.y + a.z * b.z + a.w * b.w;
    }
    float logit = acc + br[lane];

    // fp32 softmax across 32 lanes
    float m = logit;
    for (int d = 16; d > 0; d >>= 1) m = fmaxf(m, __shfl_xor(m, d, 32));
    float p = __expf(logit - m);
    float s = p;
    for (int d = 16; d > 0; d >>= 1) s += __shfl_xor(s, d, 32);
    float prob = p / s;

    atomicAdd(&probs_sum[lane], prob);

    // top-1 argmax (lowest index wins ties, matches top_k)
    float bv = prob; int bi = lane;
    for (int d = 16; d > 0; d >>= 1) {
        float ov = __shfl_xor(bv, d, 32); int oi = __shfl_xor(bi, d, 32);
        if (ov > bv || (ov == bv && oi < bi)) { bv = ov; bi = oi; }
    }
    // top-2: mask out the winner (probs >= 0, so -1 acts as -inf)
    float sv = (lane == bi) ? -1.0f : prob; int si = lane;
    for (int d = 16; d > 0; d >>= 1) {
        float ov = __shfl_xor(sv, d, 32); int oi = __shfl_xor(si, d, 32);
        if (ov > sv || (ov == sv && oi < si)) { sv = ov; si = oi; }
    }

    if (lane == 0) {
        float w1  = __expf(sv - bv);
        float inv = 1.0f / (1.0f + w1);
        idx_flat[2 * t]     = bi;  w_flat[2 * t]     = inv;
        idx_flat[2 * t + 1] = si;  w_flat[2 * t + 1] = w1 * inv;
        atomicAdd(&top1_cnt[bi], 1.0f);
    }
}

// ---------------------------------------------------------------- dispatch
__global__ __launch_bounds__(256) void moe_dispatch(
    const int* __restrict__ idx_flat, int* __restrict__ entry_pos,
    int* __restrict__ expert_entry, int* __restrict__ expert_cnt)
{
    __shared__ int wsum[8];
    __shared__ int sbase;
    int e = blockIdx.x;
    int tid = threadIdx.x, lane = tid & 31, wid = tid >> 5;
    if (tid == 0) sbase = 0;
    __syncthreads();

    for (int c0 = 0; c0 < TKE; c0 += 256) {
        int j = c0 + tid;
        int match = (idx_flat[j] == e) ? 1 : 0;
        int v = match;
        for (int d = 1; d < 32; d <<= 1) {
            int n = __shfl_up(v, d, 32);
            if (lane >= d) v += n;
        }
        if (lane == 31) wsum[wid] = v;
        __syncthreads();
        int wpre = 0, tot = 0;
        for (int w = 0; w < 8; ++w) { int q = wsum[w]; if (w < wid) wpre += q; tot += q; }
        int pos = sbase + wpre + (v - match);
        if (match) {
            int keep = pos < CAPE;
            entry_pos[j] = keep ? pos : -1;
            if (keep) expert_entry[e * CAPE + pos] = j;
        }
        __syncthreads();
        if (tid == 0) sbase += tot;
        __syncthreads();
    }
    if (tid == 0) expert_cnt[e] = (sbase < CAPE) ? sbase : CAPE;
}

// ---------------------------------------------------------------- gate/up GEMM + SwiGLU
__global__ __launch_bounds__(256) void moe_gateup(
    const float* __restrict__ x, const float* __restrict__ Wg, const float* __restrict__ Wu,
    const int* __restrict__ expert_entry, const int* __restrict__ expert_cnt,
    unsigned short* __restrict__ hid)
{
    int bid = blockIdx.x;
    int nt = bid & 15;            // 16 tiles over F
    int mt = (bid >> 4) & 3;      // 4 tiles over CAP
    int e  = bid >> 6;
    int m0 = mt * 128, n0 = nt * 128;
    int cnt = expert_cnt[e];
    if (m0 >= cnt) return;

    __shared__ __align__(32) unsigned short ldsA[8 * 512];
    __shared__ __align__(32) unsigned short ldsG[8 * 512];
    __shared__ __align__(32) unsigned short ldsU[8 * 512];

    int tid = threadIdx.x, lane = tid & 31, wid = tid >> 5;
    int laneHalf = lane >> 4, laneM = lane & 15;
    int frag = wid;

    int rowA = m0 + frag * 16 + laneM;
    const float* xrow = nullptr;
    if (rowA < cnt) {
        int j = expert_entry[e * CAPE + rowA];
        xrow = x + (size_t)(j >> 1) * HDIM;
    }
    const float* wgBase = Wg + (size_t)e * HDIM * FDIM + (size_t)(n0 + frag * 16);
    const float* wuBase = Wu + (size_t)e * HDIM * FDIM + (size_t)(n0 + frag * 16);

    unsigned int* aD32 = (unsigned int*)&ldsA[frag * 512 + lane * 16];
    unsigned int* gD32 = (unsigned int*)&ldsG[frag * 512 + lane * 16];
    unsigned int* uD32 = (unsigned int*)&ldsU[frag * 512 + lane * 16];

    int wm = wid & 3, wn = wid >> 2;

    v8f z = {0.f, 0.f, 0.f, 0.f, 0.f, 0.f, 0.f, 0.f};
    v8f accg[2][4], accu[2][4];
    for (int i = 0; i < 2; ++i)
        for (int f = 0; f < 4; ++f) { accg[i][f] = z; accu[i][f] = z; }

    for (int kk = 0; kk < HDIM; kk += 32) {
        __syncthreads();
        // --- A tile: fragment layout (lane=M; slot group selects K range) ---
        #pragma unroll
        for (int g8 = 0; g8 < 2; ++g8) {
            int kb = kk + g8 * 16 + laneHalf * 8;
            float4 v0 = {0, 0, 0, 0}, v1 = {0, 0, 0, 0};
            if (xrow) {
                const float4* p = (const float4*)(xrow + kb);
                v0 = p[0]; v1 = p[1];
            }
            aD32[g8 * 4 + 0] = pkbf(v0.x, v0.y);
            aD32[g8 * 4 + 1] = pkbf(v0.z, v0.w);
            aD32[g8 * 4 + 2] = pkbf(v1.x, v1.y);
            aD32[g8 * 4 + 3] = pkbf(v1.z, v1.w);
        }
        // --- B tiles: load K row = lane; transpose into fragment layout (lane=N; slot group selects K range) ---
        {
            const float4* gp = (const float4*)(wgBase + (size_t)(kk + lane) * FDIM);
            const float4* up = (const float4*)(wuBase + (size_t)(kk + lane) * FDIM);
            unsigned short* gS = &ldsG[frag * 512];
            unsigned short* uS = &ldsU[frag * 512];
            int kh = (lane >> 3) & 1, ks = (lane & 7) + (lane >> 4) * 8;
            #pragma unroll
            for (int q = 0; q < 4; ++q) {
                float4 gv = gp[q], uv = up[q];
                gS[(q * 4 + 0 + 16 * kh) * 16 + ks] = f2bf(gv.x);
                gS[(q * 4 + 1 + 16 * kh) * 16 + ks] = f2bf(gv.y);
                gS[(q * 4 + 2 + 16 * kh) * 16 + ks] = f2bf(gv.z);
                gS[(q * 4 + 3 + 16 * kh) * 16 + ks] = f2bf(gv.w);
                uS[(q * 4 + 0 + 16 * kh) * 16 + ks] = f2bf(uv.x);
                uS[(q * 4 + 1 + 16 * kh) * 16 + ks] = f2bf(uv.y);
                uS[(q * 4 + 2 + 16 * kh) * 16 + ks] = f2bf(uv.z);
                uS[(q * 4 + 3 + 16 * kh) * 16 + ks] = f2bf(uv.w);
            }
        }
        __syncthreads();
        v16bf a0 = *(const v16bf*)&ldsA[(wm * 2 + 0) * 512 + lane * 16];
        v16bf a1 = *(const v16bf*)&ldsA[(wm * 2 + 1) * 512 + lane * 16];
        #pragma unroll
        for (int f = 0; f < 4; ++f) {
            v16bf bg = *(const v16bf*)&ldsG[(wn * 4 + f) * 512 + lane * 16];
            accg[0][f] = wmma_bf16(a0, bg, accg[0][f]);
            accg[1][f] = wmma_bf16(a1, bg, accg[1][f]);
            v16bf bu = *(const v16bf*)&ldsU[(wn * 4 + f) * 512 + lane * 16];
            accu[0][f] = wmma_bf16(a0, bu, accu[0][f]);
            accu[1][f] = wmma_bf16(a1, bu, accu[1][f]);
        }
    }

    // --- fused SwiGLU epilogue, store hid as bf16 ---
    for (int mi = 0; mi < 2; ++mi) {
        int rowBase = m0 + (wm * 2 + mi) * 16 + 8 * laneHalf;
        for (int f = 0; f < 4; ++f) {
            int col = n0 + (wn * 4 + f) * 16 + laneM;
            #pragma unroll
            for (int i = 0; i < 8; ++i) {
                float g = accg[mi][f][i];
                float u = accu[mi][f][i];
                float hv = g / (1.0f + __expf(-g)) * u;   // silu(g) * u
                hid[((size_t)e * CAPE + (rowBase + i)) * FDIM + col] = f2bf(hv);
            }
        }
    }
}

// ---------------------------------------------------------------- down GEMM
// A-tile (hid, already bf16) staged with GLOBAL_LOAD_ASYNC_TO_LDS_B128:
// per-lane LDS destination address puts it directly in WMMA fragment order.
__global__ __launch_bounds__(256) void moe_down(
    const unsigned short* __restrict__ hid, const float* __restrict__ Wd,
    const int* __restrict__ expert_cnt, float* __restrict__ y)
{
    int bid = blockIdx.x;
    int nt = bid & 7;             // 8 tiles over H
    int mt = (bid >> 3) & 3;      // 4 tiles over CAP
    int e  = bid >> 5;
    int m0 = mt * 128, n0 = nt * 128;
    int cnt = expert_cnt[e];
    if (m0 >= cnt) return;

    __shared__ __align__(32) unsigned short ldsA[8 * 512];
    __shared__ __align__(32) unsigned short ldsB[8 * 512];

    int tid = threadIdx.x, lane = tid & 31, wid = tid >> 5;
    int laneHalf = lane >> 4, laneM = lane & 15;
    int frag = wid;

    int rowA = m0 + frag * 16 + laneM;   // always in-range; garbage rows never read back
    const unsigned short* hrow = hid + ((size_t)e * CAPE + rowA) * FDIM;
    const float* wdBase = Wd + (size_t)e * FDIM * HDIM + (size_t)(n0 + frag * 16);

    // Low 32 bits of a generic shared pointer == LDS byte offset (ISA aperture rule).
    unsigned int ldsDst = (unsigned int)(uintptr_t)&ldsA[frag * 512 + lane * 16];
    unsigned int* bD32  = (unsigned int*)&ldsB[frag * 512 + lane * 16];

    int wm = wid & 3, wn = wid >> 2;
    v8f z = {0.f, 0.f, 0.f, 0.f, 0.f, 0.f, 0.f, 0.f};
    v8f acc[2][4];
    for (int i = 0; i < 2; ++i)
        for (int f = 0; f < 4; ++f) acc[i][f] = z;

    for (int kk = 0; kk < FDIM; kk += 32) {
        __syncthreads();
        // --- A tile: async DMA copy, 16B per lane per slot-group ---
        #pragma unroll
        for (int g8 = 0; g8 < 2; ++g8) {
            int kb = kk + g8 * 16 + laneHalf * 8;
            unsigned long long src = (unsigned long long)(uintptr_t)(hrow + kb);
            asm volatile("global_load_async_to_lds_b128 %0, %1, off"
                         :: "v"(ldsDst + (unsigned)(g8 * 16)), "v"(src)
                         : "memory");
        }
        // --- B from Wd (fp32 -> bf16), load K row = lane over F; transpose into fragment layout (lane=N) ---
        {
            const float4* bp = (const float4*)(wdBase + (size_t)(kk + lane) * HDIM);
            unsigned short* bS = &ldsB[frag * 512];
            int kh = (lane >> 3) & 1, ks = (lane & 7) + (lane >> 4) * 8;
            #pragma unroll
            for (int q = 0; q < 4; ++q) {
                float4 bv = bp[q];
                bS[(q * 4 + 0 + 16 * kh) * 16 + ks] = f2bf(bv.x);
                bS[(q * 4 + 1 + 16 * kh) * 16 + ks] = f2bf(bv.y);
                bS[(q * 4 + 2 + 16 * kh) * 16 + ks] = f2bf(bv.z);
                bS[(q * 4 + 3 + 16 * kh) * 16 + ks] = f2bf(bv.w);
            }
        }
        wait_async0();
        __syncthreads();
        v16bf a0 = *(const v16bf*)&ldsA[(wm * 2 + 0) * 512 + lane * 16];
        v16bf a1 = *(const v16bf*)&ldsA[(wm * 2 + 1) * 512 + lane * 16];
        #pragma unroll
        for (int f = 0; f < 4; ++f) {
            v16bf b = *(const v16bf*)&ldsB[(wn * 4 + f) * 512 + lane * 16];
            acc[0][f] = wmma_bf16(a0, b, acc[0][f]);
            acc[1][f] = wmma_bf16(a1, b, acc[1][f]);
        }
    }

    for (int mi = 0; mi < 2; ++mi) {
        int rowBase = m0 + (wm * 2 + mi) * 16 + 8 * laneHalf;
        for (int f = 0; f < 4; ++f) {
            int col = n0 + (wn * 4 + f) * 16 + laneM;
            #pragma unroll
            for (int i = 0; i < 8; ++i)
                y[((size_t)e * CAPE + (rowBase + i)) * HDIM + col] = acc[mi][f][i];
        }
    }
}

// ---------------------------------------------------------------- combine
__global__ __launch_bounds__(256) void moe_combine(
    const int* __restrict__ idx_flat, const float* __restrict__ w_flat,
    const int* __restrict__ entry_pos, const float* __restrict__ y,
    float* __restrict__ out)
{
    int g = blockIdx.x * 256 + threadIdx.x;   // over T*H
    int t = g >> 10;
    int h = g & (HDIM - 1);
    float acc = 0.0f;
    #pragma unroll
    for (int k = 0; k < 2; ++k) {
        int j = 2 * t + k;
        int p = entry_pos[j];
        if (p >= 0) {
            int e = idx_flat[j];
            acc += w_flat[j] * y[((size_t)e * CAPE + p) * HDIM + h];
        }
    }
    out[g] = acc;
}

// ---------------------------------------------------------------- aux loss
__global__ void moe_aux(const float* __restrict__ probs_sum,
                        const float* __restrict__ top1_cnt,
                        float* __restrict__ out_aux)
{
    int lane = threadIdx.x;
    float v = (probs_sum[lane] / (float)TTOK) * (top1_cnt[lane] / (float)TTOK);
    for (int d = 16; d > 0; d >>= 1) v += __shfl_xor(v, d, 32);
    if (lane == 0) out_aux[0] = v * (float)NEXP * 0.01f;
}

// ---------------------------------------------------------------- launch
extern "C" void kernel_launch(void* const* d_in, const int* in_sizes, int n_in,
                              void* d_out, int out_size, void* d_ws, size_t ws_size,
                              hipStream_t stream)
{
    (void)in_sizes; (void)n_in; (void)out_size; (void)ws_size;
    const float* x  = (const float*)d_in[0];
    const float* Wr = (const float*)d_in[1];
    const float* br = (const float*)d_in[2];
    const float* Wg = (const float*)d_in[3];
    const float* Wu = (const float*)d_in[4];
    const float* Wd = (const float*)d_in[5];
    float* out = (float*)d_out;

    char* ws = (char*)d_ws;
    int*   idx_flat     = (int*)  (ws + 0);        // 8192 * 4
    float* w_flat       = (float*)(ws + 32768);    // 8192 * 4
    int*   entry_pos    = (int*)  (ws + 65536);    // 8192 * 4
    int*   expert_entry = (int*)  (ws + 98304);    // 32*512*4
    int*   expert_cnt   = (int*)  (ws + 163840);   // 32*4
    float* probs_sum    = (float*)(ws + 163968);   // 32*4
    float* top1_cnt     = (float*)(ws + 164096);   // 32*4 (contiguous with probs_sum)
    unsigned short* hid = (unsigned short*)(ws + (1u << 20));                         // 64 MB bf16
    float* y            = (float*)(ws + (1u << 20) + (size_t)NEXP * CAPE * FDIM * 2); // 64 MB f32

    moe_zero    <<<1, 64, 0, stream>>>(probs_sum);
    moe_router  <<<TTOK / 4, 128, 0, stream>>>(x, Wr, br, idx_flat, w_flat, probs_sum, top1_cnt);
    moe_dispatch<<<NEXP, 256, 0, stream>>>(idx_flat, entry_pos, expert_entry, expert_cnt);
    moe_gateup  <<<NEXP * 4 * 16, 256, 0, stream>>>(x, Wg, Wu, expert_entry, expert_cnt, hid);
    moe_down    <<<NEXP * 4 * 8, 256, 0, stream>>>(hid, Wd, expert_cnt, y);
    moe_combine <<<(TTOK * HDIM) / 256, 256, 0, stream>>>(idx_flat, w_flat, entry_pos, y, out);
    moe_aux     <<<1, 32, 0, stream>>>(probs_sum, top1_cnt, out + (size_t)TTOK * HDIM);
}
